// MultiHeadAttentionRoPE_33457795235983
// MI455X (gfx1250) — matrix operand order repaired and checked by reading, not compiled, not run on hardware
//
#include <hip/hip_runtime.h>

// ---------------- types ----------------
typedef unsigned short u16;
typedef unsigned int   u32;
typedef __attribute__((ext_vector_type(16))) __bf16 v16bf;
typedef __attribute__((ext_vector_type(8)))  float  v8f;
typedef __attribute__((ext_vector_type(4)))  u32    u32x4;
typedef __attribute__((ext_vector_type(4)))  float  f32x4;
typedef __attribute__((ext_vector_type(4)))  u16    u16x4;

union FragBF {            // one 16x32 A-frag or 32x16 B-frag (bf16), 32B/lane
  v16bf v;
  u16   u[16];
  u32   w[8];
  u32x4 q[2];
};

__device__ __forceinline__ u16 f2bf(float f) {
  union { float f; u32 u; } x; x.f = f;
  u32 u = x.u;
  return (u16)((u + 0x7FFFu + ((u >> 16) & 1u)) >> 16);  // RNE
}

#define WMMA_BF16(A, B, C) \
  __builtin_amdgcn_wmma_f32_16x16x32_bf16(false, (A), false, (B), (short)0, (C), false, false)

// problem constants
#define BATCH  2
#define SEQ    2048
#define DM     1024
#define NH     16
#define DH     64

// ---------------- 1) fp32 -> bf16 convert (x) ----------------
__global__ __launch_bounds__(256) void k_cvt_x(const float* __restrict__ x,
                                               u16* __restrict__ o, int n4) {
  int i = (blockIdx.x * 256 + threadIdx.x);
  if (i >= n4) return;
  f32x4 v = *(const f32x4*)(x + i * 4);
  u16x4 r; r.x = f2bf(v.x); r.y = f2bf(v.y); r.z = f2bf(v.z); r.w = f2bf(v.w);
  *(u16x4*)(o + i * 4) = r;
}

// ---------------- 2) transpose-convert weights: Wt[n][k] = bf16(W[k][n]) ----
__global__ __launch_bounds__(256) void k_wtrans(const float* __restrict__ w0,
                                                const float* __restrict__ w1,
                                                const float* __restrict__ w2,
                                                const float* __restrict__ w3,
                                                u16* __restrict__ o0, u16* __restrict__ o1,
                                                u16* __restrict__ o2, u16* __restrict__ o3) {
  const float* W; u16* O;
  switch (blockIdx.z) {
    case 0: W = w0; O = o0; break;
    case 1: W = w1; O = o1; break;
    case 2: W = w2; O = o2; break;
    default: W = w3; O = o3; break;
  }
  __shared__ u16 tile[32][33];
  int tx = threadIdx.x, ty = threadIdx.y;          // 32 x 8
  int n0 = blockIdx.x * 32, k0 = blockIdx.y * 32;
#pragma unroll
  for (int j = 0; j < 4; ++j) {                    // read: coalesced along n
    int kl = ty + j * 8;
    tile[kl][tx] = f2bf(W[(size_t)(k0 + kl) * DM + n0 + tx]);
  }
  __syncthreads();
#pragma unroll
  for (int j = 0; j < 4; ++j) {                    // write: coalesced along k
    int nl = ty + j * 8;
    O[(size_t)(n0 + nl) * DM + k0 + tx] = tile[tx][nl];
  }
}

// ---------------- 3) fused QKV GEMM + bias + RoPE --------------------------
// out rows = B*S (4096), cols = DM (1024). Block tile 128x128, 8 waves,
// each wave: 2 M-tiles x 4 N-tiles of 16x16, K chunks of 32 (bf16 WMMA).
__global__ __launch_bounds__(256) void k_gemm_qkv(
    const u16* __restrict__ Xbf,
    const u16* __restrict__ Wqt, const u16* __restrict__ Wkt, const u16* __restrict__ Wvt,
    const float* __restrict__ bq, const float* __restrict__ bk, const float* __restrict__ bv,
    u16* __restrict__ Qo, u16* __restrict__ Ko, u16* __restrict__ Vo) {
  const int z = blockIdx.z;                        // 0=Q 1=K 2=V
  const u16*  Wt   = (z == 0) ? Wqt : (z == 1) ? Wkt : Wvt;
  const float* bias = (z == 0) ? bq : (z == 1) ? bk : bv;

  const int lane = threadIdx.x & 31;
  const int wv   = threadIdx.x >> 5;               // 0..7
  const int hi   = lane >> 4;                      // half-wave select
  const int ln   = lane & 15;
  const int rb = blockIdx.x * 128 + (wv >> 1) * 32;   // wave row base
  const int cb = blockIdx.y * 128 + (wv & 1) * 64;    // wave col base

  v8f acc[2][4];
  const v8f vzero = {0.f, 0.f, 0.f, 0.f, 0.f, 0.f, 0.f, 0.f};
#pragma unroll
  for (int mi = 0; mi < 2; ++mi)
#pragma unroll
    for (int ni = 0; ni < 4; ++ni) acc[mi][ni] = vzero;

  for (int k0 = 0; k0 < DM; k0 += 32) {
    FragBF a[2], b[4];
#pragma unroll
    for (int mi = 0; mi < 2; ++mi) {               // A: 16x32, split K chunks
      const u16* p = Xbf + (size_t)(rb + mi * 16 + ln) * DM + k0 + 8 * hi;
      a[mi].q[0] = *(const u32x4*)p;
      a[mi].q[1] = *(const u32x4*)(p + 16);
    }
#pragma unroll
    for (int ni = 0; ni < 4; ++ni) {               // B: 32x16, split K chunks
      const u16* p = Wt + (size_t)(cb + ni * 16 + ln) * DM + k0 + 8 * hi;
      b[ni].q[0] = *(const u32x4*)p;
      b[ni].q[1] = *(const u32x4*)(p + 16);
    }
#pragma unroll
    for (int mi = 0; mi < 2; ++mi)
#pragma unroll
      for (int ni = 0; ni < 4; ++ni)
        acc[mi][ni] = WMMA_BF16(a[mi].v, b[ni].v, acc[mi][ni]);
  }

  // epilogue: bias (+RoPE for Q,K), scatter bf16 into head layouts
#pragma unroll
  for (int ni = 0; ni < 4; ++ni) {
    const int n = cb + ni * 16 + ln;               // model column
    const float bb = bias[n];
    const int h = n >> 6, d = n & 63, j = d >> 1;
    const bool evn = (d & 1) == 0;
    const float invf = __powf(10000.0f, -(float)(2 * j) * (1.0f / 64.0f));
#pragma unroll
    for (int mi = 0; mi < 2; ++mi) {
#pragma unroll
      for (int e = 0; e < 8; ++e) {
        const int r = rb + mi * 16 + 8 * hi + e;   // global row (b*S + s)
        const int bi = r >> 11, spos = r & (SEQ - 1);
        float val = acc[mi][ni][e] + bb;
        if (z < 2) {                               // RoPE via lane-pair shuffle
          float sn, cs;
          __sincosf((float)spos * invf, &sn, &cs);
          float pv = __shfl_xor(val, 1, 32);
          val = evn ? (val * cs - pv * sn) : (pv * sn + val * cs);
        }
        const u16 ob = f2bf(val);
        if (z == 0)
          Qo[((size_t)(bi * NH + h) * SEQ + spos) * DH + d] = ob;
        else if (z == 1)
          Ko[((size_t)(bi * NH + h) * SEQ + spos) * DH + d] = ob;
        else                                       // V transposed: [B,H,d,S]
          Vo[((size_t)(bi * NH + h) * DH + d) * SEQ + spos] = ob;
      }
    }
  }
}

// ---------------- 4) causal flash attention (bf16 WMMA, fp32 online softmax)
// One wave per 16-query tile; 32 keys per iteration.
__global__ __launch_bounds__(128) void k_attn(const u16* __restrict__ Q,
                                              const u16* __restrict__ K,
                                              const u16* __restrict__ V,
                                              u16* __restrict__ O) {
  __shared__ __align__(16) u32 pbuf[4][256];       // per-wave P tile (bf16 pairs)
  const int lane = threadIdx.x & 31;
  const int wv   = threadIdx.x >> 5;
  const int hi   = lane >> 4;
  const int ln   = lane & 15;
  const int bh = blockIdx.x;                       // 0..31
  const int q0 = (blockIdx.y * 4 + wv) * 16;
  const int b = bh >> 4, h = bh & 15;

  const u16* Qb = Q + (size_t)bh * SEQ * DH;
  const u16* Kb = K + (size_t)bh * SEQ * DH;
  const u16* Vb = V + (size_t)bh * DH * SEQ;

  FragBF qa[2];
#pragma unroll
  for (int c = 0; c < 2; ++c) {                    // Q A-frags: d chunks of 32
    const u16* p = Qb + (size_t)(q0 + ln) * DH + 32 * c + 8 * hi;
    qa[c].q[0] = *(const u32x4*)p;
    qa[c].q[1] = *(const u32x4*)(p + 16);
  }

  const v8f vzero = {0.f, 0.f, 0.f, 0.f, 0.f, 0.f, 0.f, 0.f};
  v8f acc[4] = {vzero, vzero, vzero, vzero};       // 16 q-rows x 64 d
  float rm[8], rs[8];
#pragma unroll
  for (int e = 0; e < 8; ++e) { rm[e] = -3.0e38f; rs[e] = 0.0f; }

  for (int kv0 = 0; kv0 < q0 + 16; kv0 += 32) {
    FragBF kb[2][2];
#pragma unroll
    for (int t = 0; t < 2; ++t)
#pragma unroll
      for (int c = 0; c < 2; ++c) {                // K^T B-frags
        const u16* p = Kb + (size_t)(kv0 + 16 * t + ln) * DH + 32 * c + 8 * hi;
        kb[t][c].q[0] = *(const u32x4*)p;
        kb[t][c].q[1] = *(const u32x4*)(p + 16);
      }
    v8f s0 = WMMA_BF16(qa[0].v, kb[0][0].v, vzero);
    s0     = WMMA_BF16(qa[1].v, kb[0][1].v, s0);
    v8f s1 = WMMA_BF16(qa[0].v, kb[1][0].v, vzero);
    s1     = WMMA_BF16(qa[1].v, kb[1][1].v, s1);

#pragma unroll
    for (int e = 0; e < 8; ++e) {
      const int row = q0 + 8 * hi + e;
      const int c0 = kv0 + ln, c1 = c0 + 16;
      float v0 = s0[e] * 0.125f; if (c0 > row) v0 = -3.0e38f;
      float v1 = s1[e] * 0.125f; if (c1 > row) v1 = -3.0e38f;
      float mx = fmaxf(v0, v1);                    // row max over 16 lanes/half
      mx = fmaxf(mx, __shfl_xor(mx, 1, 32));
      mx = fmaxf(mx, __shfl_xor(mx, 2, 32));
      mx = fmaxf(mx, __shfl_xor(mx, 4, 32));
      mx = fmaxf(mx, __shfl_xor(mx, 8, 32));
      const float mnew = fmaxf(rm[e], mx);
      const float corr = __expf(rm[e] - mnew);
      rm[e] = mnew;
      const float p0 = __expf(v0 - mnew);
      const float p1 = __expf(v1 - mnew);
      float ps = p0 + p1;
      ps += __shfl_xor(ps, 1, 32);
      ps += __shfl_xor(ps, 2, 32);
      ps += __shfl_xor(ps, 4, 32);
      ps += __shfl_xor(ps, 8, 32);
      rs[e] = rs[e] * corr + ps;
#pragma unroll
      for (int t = 0; t < 4; ++t) acc[t][e] *= corr;
      // C-layout -> LDS (packed bf16 pair: tile0|tile1)
      pbuf[wv][(8 * hi + e) * 16 + ln] = (u32)f2bf(p0) | ((u32)f2bf(p1) << 16);
    }

    // read back P in A-fragment layout (same wave; LDS is in-order)
    FragBF raw, pa;
    const u32x4* pp = (const u32x4*)&pbuf[wv][ln * 16 + 8 * hi];
    raw.q[0] = pp[0];
    raw.q[1] = pp[1];
#pragma unroll
    for (int i = 0; i < 8; ++i) {                  // de-interleave pairs
      pa.u[i]     = (u16)(raw.w[i] & 0xFFFFu);     // K = 8*hi+i      (tile0)
      pa.u[8 + i] = (u16)(raw.w[i] >> 16);         // K = 16+8*hi+i   (tile1)
    }

#pragma unroll
    for (int t = 0; t < 4; ++t) {                  // P(16x32) @ V(32x16) per d-tile
      FragBF vb;
      const u16* p = Vb + (size_t)(16 * t + ln) * SEQ + kv0 + 8 * hi;
      vb.q[0] = *(const u32x4*)p;
      vb.q[1] = *(const u32x4*)(p + 16);
      acc[t] = WMMA_BF16(pa.v, vb.v, acc[t]);
    }
  }

  // epilogue: normalize, write bf16 O[B,S,DM] (col = h*64 + d)
#pragma unroll
  for (int t = 0; t < 4; ++t)
#pragma unroll
    for (int e = 0; e < 8; ++e) {
      const int row = q0 + 8 * hi + e;
      const float val = acc[t][e] / rs[e];
      O[(size_t)(b * SEQ + row) * DM + h * DH + 16 * t + ln] = f2bf(val);
    }
}

// ---------------- 5) output projection GEMM + bias (fp32 out) --------------
__global__ __launch_bounds__(256) void k_gemm_out(const u16* __restrict__ Abf,
                                                  const u16* __restrict__ Wt,
                                                  const float* __restrict__ bias,
                                                  float* __restrict__ out) {
  const int lane = threadIdx.x & 31;
  const int wv   = threadIdx.x >> 5;
  const int hi   = lane >> 4;
  const int ln   = lane & 15;
  const int rb = blockIdx.x * 128 + (wv >> 1) * 32;
  const int cb = blockIdx.y * 128 + (wv & 1) * 64;

  v8f acc[2][4];
  const v8f vzero = {0.f, 0.f, 0.f, 0.f, 0.f, 0.f, 0.f, 0.f};
#pragma unroll
  for (int mi = 0; mi < 2; ++mi)
#pragma unroll
    for (int ni = 0; ni < 4; ++ni) acc[mi][ni] = vzero;

  for (int k0 = 0; k0 < DM; k0 += 32) {
    FragBF a[2], b[4];
#pragma unroll
    for (int mi = 0; mi < 2; ++mi) {
      const u16* p = Abf + (size_t)(rb + mi * 16 + ln) * DM + k0 + 8 * hi;
      a[mi].q[0] = *(const u32x4*)p;
      a[mi].q[1] = *(const u32x4*)(p + 16);
    }
#pragma unroll
    for (int ni = 0; ni < 4; ++ni) {
      const u16* p = Wt + (size_t)(cb + ni * 16 + ln) * DM + k0 + 8 * hi;
      b[ni].q[0] = *(const u32x4*)p;
      b[ni].q[1] = *(const u32x4*)(p + 16);
    }
#pragma unroll
    for (int mi = 0; mi < 2; ++mi)
#pragma unroll
      for (int ni = 0; ni < 4; ++ni)
        acc[mi][ni] = WMMA_BF16(a[mi].v, b[ni].v, acc[mi][ni]);
  }

#pragma unroll
  for (int ni = 0; ni < 4; ++ni) {
    const int n = cb + ni * 16 + ln;
    const float bb = bias[n];
#pragma unroll
    for (int mi = 0; mi < 2; ++mi)
#pragma unroll
      for (int e = 0; e < 8; ++e) {
        const int r = rb + mi * 16 + 8 * hi + e;
        out[(size_t)r * DM + n] = acc[mi][ni][e] + bb;   // coalesced across lanes
      }
  }
}

// ---------------- launch ----------------
extern "C" void kernel_launch(void* const* d_in, const int* in_sizes, int n_in,
                              void* d_out, int out_size, void* d_ws, size_t ws_size,
                              hipStream_t stream) {
  const float* x  = (const float*)d_in[0];
  const float* Wq = (const float*)d_in[1];
  const float* bq = (const float*)d_in[2];
  const float* Wk = (const float*)d_in[3];
  const float* bk = (const float*)d_in[4];
  const float* Wv = (const float*)d_in[5];
  const float* bv = (const float*)d_in[6];
  const float* Wo = (const float*)d_in[7];
  const float* bo = (const float*)d_in[8];

  const size_t MB = 1024ull * 1024ull;
  if (ws_size < 48 * MB) return;                   // workspace budget (48 MB)
  char* ws = (char*)d_ws;
  u16* Xbf = (u16*)(ws + 0 * MB);                  // 8 MB  [4096][1024]
  u16* Wqt = (u16*)(ws + 8 * MB);                  // 2 MB  [n][k]
  u16* Wkt = (u16*)(ws + 10 * MB);
  u16* Wvt = (u16*)(ws + 12 * MB);
  u16* Wot = (u16*)(ws + 14 * MB);
  u16* Qbf = (u16*)(ws + 16 * MB);                 // 8 MB  [B,H,S,d]
  u16* Kbf = (u16*)(ws + 24 * MB);                 // 8 MB  [B,H,S,d]
  u16* Vtb = (u16*)(ws + 32 * MB);                 // 8 MB  [B,H,d,S]
  u16* Obf = (u16*)(ws + 40 * MB);                 // 8 MB  [4096][1024]

  // 1) convert x
  const int n4 = (BATCH * SEQ * DM) / 4;
  k_cvt_x<<<(n4 + 255) / 256, 256, 0, stream>>>(x, Xbf, n4);

  // 2) transpose-convert weights
  k_wtrans<<<dim3(DM / 32, DM / 32, 4), dim3(32, 8), 0, stream>>>(
      Wq, Wk, Wv, Wo, Wqt, Wkt, Wvt, Wot);

  // 3) QKV projections + bias + RoPE
  k_gemm_qkv<<<dim3((BATCH * SEQ) / 128, DM / 128, 3), 256, 0, stream>>>(
      Xbf, Wqt, Wkt, Wvt, bq, bk, bv, Qbf, Kbf, Vtb);

  // 4) causal flash attention
  k_attn<<<dim3(BATCH * NH, SEQ / (4 * 16)), 128, 0, stream>>>(Qbf, Kbf, Vtb, Obf);

  // 5) output projection
  k_gemm_out<<<dim3((BATCH * SEQ) / 128, DM / 128), 256, 0, stream>>>(
      Obf, Wot, bo, (float*)d_out);
}
